// SwinBlock_7550552506834
// MI455X (gfx1250) — matrix operand order repaired and checked by reading, not compiled, not run on hardware
//
#include <hip/hip_runtime.h>
#include <hip/hip_bf16.h>

typedef __attribute__((ext_vector_type(16))) _Float16 v16h;
typedef __attribute__((ext_vector_type(8)))  _Float16 v8h;
typedef __attribute__((ext_vector_type(4)))  _Float16 v4h;
typedef __attribute__((ext_vector_type(8)))  float    v8f;

#define BATCH   64
#define HH      28
#define WW      28
#define CCH     512
#define NHEADS  16
#define HDIM    32
#define WIN     7
#define NTOK    49
#define NWIN    16
#define BW      (BATCH * NWIN)         // 1024 windows
#define MTOK    (BATCH * HH * WW)      // 50176 tokens
#define MLPHID  2048

// ---------------------------------------------------------------------------
// helpers
// ---------------------------------------------------------------------------
__device__ __forceinline__ int win_to_orig(int t) {
  // windowed+shifted token index -> original token index (B, H*W)
  int b   = t / 784;
  int rem = t - b * 784;
  int win = rem / 49;
  int pos = rem - win * 49;
  int hb = win >> 2, wb = win & 3;
  int i = pos / 7, j = pos - i * 7;
  int h = hb * 7 + i + 3; if (h >= 28) h -= 28;
  int w = wb * 7 + j + 3; if (w >= 28) w -= 28;
  return b * 784 + h * 28 + w;
}

__device__ __forceinline__ float gelu_tanh(float x) {
  const float k0 = 0.7978845608028654f;   // sqrt(2/pi)
  float x3 = x * x * x;
  return 0.5f * x * (1.0f + tanhf(k0 * (x + 0.044715f * x3)));
}

// A fragment: 16x32 f16, row-major source [row][k], pitch ld (halfs).
// lane%16 = row; halfs 0..7 -> k0+(lane/16)*8.., halfs 8..15 -> k0+16+(lane/16)*8..
__device__ __forceinline__ v16h load_a_frag(const _Float16* base, int row0,
                                            int k0, int ld, int lane) {
  int r  = lane & 15;
  int kg = lane >> 4;
  const _Float16* p = base + (size_t)(row0 + r) * ld + k0 + kg * 8;
  v8h lo = *(const v8h*)p;
  v8h hi = *(const v8h*)(p + 16);
  return __builtin_shufflevector(lo, hi, 0, 1, 2, 3, 4, 5, 6, 7,
                                 8, 9, 10, 11, 12, 13, 14, 15);
}

// B fragment: 32x16 f16 from B^T storage [n][k], pitch ld (halfs).
// lane%16 = column n; halfs 0..7 -> k0+(lane/16)*8.., halfs 8..15 -> k0+16+(lane/16)*8..
__device__ __forceinline__ v16h load_b_frag(const _Float16* base, int col0,
                                            int k0, int ld, int lane) {
  int r  = lane & 15;
  int kg = lane >> 4;
  const _Float16* p = base + (size_t)(col0 + r) * ld + k0 + kg * 8;
  v8h lo = *(const v8h*)p;
  v8h hi = *(const v8h*)(p + 16);
  return __builtin_shufflevector(lo, hi, 0, 1, 2, 3, 4, 5, 6, 7,
                                 8, 9, 10, 11, 12, 13, 14, 15);
}

__device__ __forceinline__ v8f wmma_f16(v16h a, v16h b, v8f c) {
  return __builtin_amdgcn_wmma_f32_16x16x32_f16(false, a, false, b,
                                                (short)0, c, false, false);
}

// ---------------------------------------------------------------------------
// weight convert + transpose: src f32 [K][N] -> dst f16 [N][K]
// ---------------------------------------------------------------------------
__global__ void cvt_transpose(const float* __restrict__ src,
                              _Float16* __restrict__ dst, int K, int N) {
  int idx = blockIdx.x * 256 + threadIdx.x;
  if (idx < K * N) {
    int k = idx / N;
    int n = idx - k * N;
    dst[(size_t)n * K + k] = (_Float16)src[idx];
  }
}

// ---------------------------------------------------------------------------
// LayerNorm (one wave32 per token), optional shift+window permutation on read
// ---------------------------------------------------------------------------
template <bool PERM>
__global__ __launch_bounds__(256)
void ln_kernel(const float* __restrict__ src, const float* __restrict__ gamma,
               const float* __restrict__ beta, _Float16* __restrict__ dst) {
  int wave = threadIdx.x >> 5;
  int lane = threadIdx.x & 31;
  int t = blockIdx.x * 8 + wave;           // destination (windowed) token
  int o = PERM ? win_to_orig(t) : t;       // source token
  const float* row = src + (size_t)o * CCH;

  float4 v[4];
  float sum = 0.f, ss = 0.f;
#pragma unroll
  for (int kq = 0; kq < 4; ++kq) {
    v[kq] = *(const float4*)(row + kq * 128 + lane * 4);
    sum += v[kq].x + v[kq].y + v[kq].z + v[kq].w;
    ss  += v[kq].x * v[kq].x + v[kq].y * v[kq].y +
           v[kq].z * v[kq].z + v[kq].w * v[kq].w;
  }
#pragma unroll
  for (int off = 16; off > 0; off >>= 1) {
    sum += __shfl_xor(sum, off, 32);
    ss  += __shfl_xor(ss, off, 32);
  }
  float mu   = sum * (1.0f / CCH);
  float var  = ss * (1.0f / CCH) - mu * mu;
  float rstd = rsqrtf(var + 1e-5f);

  _Float16* drow = dst + (size_t)t * CCH;
#pragma unroll
  for (int kq = 0; kq < 4; ++kq) {
    int c = kq * 128 + lane * 4;
    float4 g = *(const float4*)(gamma + c);
    float4 b = *(const float4*)(beta + c);
    v4h oz;
    oz[0] = (_Float16)((v[kq].x - mu) * rstd * g.x + b.x);
    oz[1] = (_Float16)((v[kq].y - mu) * rstd * g.y + b.y);
    oz[2] = (_Float16)((v[kq].z - mu) * rstd * g.z + b.z);
    oz[3] = (_Float16)((v[kq].w - mu) * rstd * g.w + b.w);
    *(v4h*)(drow + c) = oz;
  }
}

// ---------------------------------------------------------------------------
// Generic WMMA GEMM: Out[M,N] = epi( X[M,K] @ Wt[N,K]^T + bias[N] )
// block = 128 threads (4 waves, 2x2), block tile 128(M) x 128(N), wave 64x64.
// 8 fragment loads feed 16 WMMAs per K-step (0.5 load/WMMA).
// EPI: 0 = f16 store, 1 = gelu->f16, 2 = +resid f32 with window-reverse perm,
//      3 = +resid f32 identity (resid may alias outf)
// ---------------------------------------------------------------------------
template <int EPI>
__global__ __launch_bounds__(128)
void wmma_gemm(const _Float16* __restrict__ X, const _Float16* __restrict__ Wt,
               const float* __restrict__ bias, _Float16* __restrict__ outh,
               float* __restrict__ outf, const float* __restrict__ resid,
               int M, int Nn, int K) {
  const int lane = threadIdx.x & 31;
  const int wv   = threadIdx.x >> 5;
  const int wm   = wv >> 1, wn = wv & 1;
  const int M0   = blockIdx.y * 128 + wm * 64;
  const int N0   = blockIdx.x * 128 + wn * 64;

  v8f acc[4][4];
  const v8f zero = {0.f, 0.f, 0.f, 0.f, 0.f, 0.f, 0.f, 0.f};
#pragma unroll
  for (int s = 0; s < 4; ++s)
#pragma unroll
    for (int t = 0; t < 4; ++t) acc[s][t] = zero;

  for (int kk = 0; kk < K; kk += 32) {
    if (kk + 32 < K) {
      // hint next K-slab into cache (lowers to global_prefetch_b8)
      __builtin_prefetch(X + (size_t)(M0 + lane) * K + kk + 32, 0, 0);
      __builtin_prefetch(X + (size_t)(M0 + 32 + lane) * K + kk + 32, 0, 0);
      __builtin_prefetch(Wt + (size_t)(N0 + lane) * K + kk + 32, 0, 0);
      __builtin_prefetch(Wt + (size_t)(N0 + 32 + lane) * K + kk + 32, 0, 0);
    }
    v16h a[4], b[4];
#pragma unroll
    for (int s = 0; s < 4; ++s) a[s] = load_a_frag(X, M0 + 16 * s, kk, K, lane);
#pragma unroll
    for (int t = 0; t < 4; ++t) b[t] = load_b_frag(Wt, N0 + 16 * t, kk, K, lane);
#pragma unroll
    for (int s = 0; s < 4; ++s)
#pragma unroll
      for (int t = 0; t < 4; ++t) acc[s][t] = wmma_f16(a[s], b[t], acc[s][t]);
  }

  const int mloc = (lane >> 4) << 3;
  const int nloc = lane & 15;
  float bv[4];
#pragma unroll
  for (int t = 0; t < 4; ++t) bv[t] = bias[N0 + 16 * t + nloc];
#pragma unroll
  for (int s = 0; s < 4; ++s) {
#pragma unroll
    for (int r = 0; r < 8; ++r) {
      int m = M0 + 16 * s + mloc + r;
      size_t base;
      if (EPI == 2) {
        base = (size_t)win_to_orig(m) * Nn;
      } else {
        base = (size_t)m * Nn;
      }
#pragma unroll
      for (int t = 0; t < 4; ++t) {
        int n = N0 + 16 * t + nloc;
        float v = acc[s][t][r] + bv[t];
        if (EPI == 0) {
          outh[base + n] = (_Float16)v;
        } else if (EPI == 1) {
          outh[base + n] = (_Float16)gelu_tanh(v);
        } else {
          outf[base + n] = resid[base + n] + v;
        }
      }
    }
  }
}

// ---------------------------------------------------------------------------
// Windowed attention: one block per (window, head). 49 tokens padded to 64.
// ---------------------------------------------------------------------------
__device__ __forceinline__ int reg7(int x) { return x < 21 ? 0 : (x < 25 ? 1 : 2); }

__global__ __launch_bounds__(128)
void attention_kernel(const _Float16* __restrict__ qkv,
                      const float* __restrict__ rpbt,
                      _Float16* __restrict__ attout) {
  constexpr int QP = 40;   // pitch (halfs) for q_s / k_s  [64][32]
  constexpr int VP = 72;   // pitch for vt_s [32][64]
  constexpr int PP = 72;   // pitch for p_s  [64][64]
  constexpr int SP = 65;   // pitch (floats) for s_s [64][64]
  __shared__ _Float16 q_s[64 * QP];
  __shared__ _Float16 k_s[64 * QP];
  __shared__ _Float16 vt_s[32 * VP];
  __shared__ _Float16 p_s[64 * PP];
  __shared__ float    s_s[64 * SP];

  const int head = blockIdx.x;
  const int widx = blockIdx.y;               // 0..1023
  const int hb   = (widx & 15) >> 2;
  const int wb   = widx & 3;
  const size_t tokbase = (size_t)widx * 49;
  const int tid = threadIdx.x, lane = tid & 31, wv = tid >> 5;

  // ---- stage q, k, v^T into LDS (rows >= 49 zero-padded) ----
  const v8h zh = {(_Float16)0, (_Float16)0, (_Float16)0, (_Float16)0,
                  (_Float16)0, (_Float16)0, (_Float16)0, (_Float16)0};
  if (tid < 64) {
    int j = tid;
    if (j < 49) {
      const _Float16* qr = qkv + (tokbase + j) * 1536 + head * 32;
      const _Float16* kr = qr + 512;
#pragma unroll
      for (int c = 0; c < 32; c += 8) {
        *(v8h*)&q_s[j * QP + c] = *(const v8h*)(qr + c);
        *(v8h*)&k_s[j * QP + c] = *(const v8h*)(kr + c);
      }
    } else {
#pragma unroll
      for (int c = 0; c < 32; c += 8) {
        *(v8h*)&q_s[j * QP + c] = zh;
        *(v8h*)&k_s[j * QP + c] = zh;
      }
    }
  } else {
    int j = tid - 64;
    if (j < 49) {
      const _Float16* vr = qkv + (tokbase + j) * 1536 + 1024 + head * 32;
#pragma unroll
      for (int c = 0; c < 32; ++c) vt_s[c * VP + j] = vr[c];
    } else {
#pragma unroll
      for (int c = 0; c < 32; ++c) vt_s[c * VP + j] = (_Float16)0;
    }
  }
  __syncthreads();

  // ---- S = q @ k^T  (wave wv owns rows 16*wv .. 16*wv+15) ----
  const v8f zero = {0.f, 0.f, 0.f, 0.f, 0.f, 0.f, 0.f, 0.f};
  v16h aq = load_a_frag(q_s, 16 * wv, 0, QP, lane);
  v8f sacc[4];
#pragma unroll
  for (int f = 0; f < 4; ++f) {
    v16h bk = load_b_frag(k_s, 16 * f, 0, QP, lane);
    sacc[f] = wmma_f16(aq, bk, zero);
  }

  // ---- scale + rel-pos bias + shift mask -> s_s (f32) ----
  const float scale = 0.17677669529663687f;   // 1/sqrt(32)
  const int mloc = (lane >> 4) << 3;
  const int nloc = lane & 15;
#pragma unroll
  for (int f = 0; f < 4; ++f) {
#pragma unroll
    for (int r = 0; r < 8; ++r) {
      int m = 16 * wv + mloc + r;
      int n = 16 * f + nloc;
      float sv;
      if (m < 49 && n < 49) {
        int mi = m / 7, mj = m - mi * 7;
        int ni = n / 7, nj = n - ni * 7;
        int ridx = (mi - ni + 6) * 13 + (mj - nj + 6);
        float bias = rpbt[ridx * NHEADS + head];
        int rm = 3 * reg7(hb * 7 + mi) + reg7(wb * 7 + mj);
        int rn = 3 * reg7(hb * 7 + ni) + reg7(wb * 7 + nj);
        float mask = (rm != rn) ? -100.0f : 0.0f;
        sv = sacc[f][r] * scale + bias + mask;
      } else {
        sv = (m >= 49 && n == 0) ? 0.0f : -1e30f;  // pad rows: delta at col 0
      }
      s_s[m * SP + n] = sv;
    }
  }
  __syncthreads();

  // ---- softmax over rows (threads 0..63, one row each) ----
  if (tid < 64) {
    int m = tid;
    float mx = -1e30f;
    for (int n = 0; n < 64; ++n) mx = fmaxf(mx, s_s[m * SP + n]);
    float sum = 0.f;
    for (int n = 0; n < 64; ++n) {
      float e = __expf(s_s[m * SP + n] - mx);
      s_s[m * SP + n] = e;
      sum += e;
    }
    float inv = 1.0f / sum;
    for (int n = 0; n < 64; ++n)
      p_s[m * PP + n] = (_Float16)(s_s[m * SP + n] * inv);
  }
  __syncthreads();

  // ---- O = P @ V  (K = 64 in two 32-steps, N = 32 channels) ----
  v8f oacc[2] = {zero, zero};
  for (int k0 = 0; k0 < 64; k0 += 32) {
    v16h ap = load_a_frag(p_s, 16 * wv, k0, PP, lane);
#pragma unroll
    for (int t = 0; t < 2; ++t) {
      v16h bv = load_b_frag(vt_s, 16 * t, k0, VP, lane);
      oacc[t] = wmma_f16(ap, bv, oacc[t]);
    }
  }
#pragma unroll
  for (int t = 0; t < 2; ++t) {
#pragma unroll
    for (int r = 0; r < 8; ++r) {
      int m = 16 * wv + mloc + r;
      if (m < 49) {
        int n = 16 * t + nloc;
        attout[(tokbase + m) * CCH + head * HDIM + n] = (_Float16)oacc[t][r];
      }
    }
  }
}

// ---------------------------------------------------------------------------
// launch
// ---------------------------------------------------------------------------
extern "C" void kernel_launch(void* const* d_in, const int* in_sizes, int n_in,
                              void* d_out, int out_size, void* d_ws,
                              size_t ws_size, hipStream_t stream) {
  const float* x       = (const float*)d_in[0];
  const float* qkv_w   = (const float*)d_in[1];
  const float* qkv_b   = (const float*)d_in[2];
  const float* proj_w  = (const float*)d_in[3];
  const float* proj_b  = (const float*)d_in[4];
  const float* rpbt    = (const float*)d_in[5];
  const float* norm1_s = (const float*)d_in[6];
  const float* norm1_b = (const float*)d_in[7];
  const float* norm2_s = (const float*)d_in[8];
  const float* norm2_b = (const float*)d_in[9];
  const float* fc1_w   = (const float*)d_in[10];
  const float* fc1_b   = (const float*)d_in[11];
  const float* fc2_w   = (const float*)d_in[12];
  const float* fc2_b   = (const float*)d_in[13];
  float* out = (float*)d_out;

  char* ws = (char*)d_ws;
  auto alloc = [&](size_t bytes) {
    void* p = (void*)ws;
    ws += (bytes + 255) & ~(size_t)255;
    return p;
  };
  _Float16* wqkvT = (_Float16*)alloc((size_t)CCH * 3 * CCH * 2);
  _Float16* wprojT = (_Float16*)alloc((size_t)CCH * CCH * 2);
  _Float16* wfc1T = (_Float16*)alloc((size_t)CCH * MLPHID * 2);
  _Float16* wfc2T = (_Float16*)alloc((size_t)MLPHID * CCH * 2);
  _Float16* Xw   = (_Float16*)alloc((size_t)MTOK * CCH * 2);
  _Float16* QKV  = (_Float16*)alloc((size_t)MTOK * 3 * CCH * 2);
  _Float16* AttO = (_Float16*)alloc((size_t)MTOK * CCH * 2);
  _Float16* H2   = (_Float16*)alloc((size_t)MTOK * CCH * 2);
  _Float16* Hh   = (_Float16*)alloc((size_t)MTOK * MLPHID * 2);

  // weights -> f16 transposed [N][K]
  cvt_transpose<<<(CCH * 3 * CCH + 255) / 256, 256, 0, stream>>>(qkv_w, wqkvT, CCH, 3 * CCH);
  cvt_transpose<<<(CCH * CCH + 255) / 256, 256, 0, stream>>>(proj_w, wprojT, CCH, CCH);
  cvt_transpose<<<(CCH * MLPHID + 255) / 256, 256, 0, stream>>>(fc1_w, wfc1T, CCH, MLPHID);
  cvt_transpose<<<(MLPHID * CCH + 255) / 256, 256, 0, stream>>>(fc2_w, wfc2T, MLPHID, CCH);

  // LN1 + shift + window partition -> Xw (f16, windowed token order)
  ln_kernel<true><<<MTOK / 8, 256, 0, stream>>>(x, norm1_s, norm1_b, Xw);

  // QKV = Xw @ Wqkv + b  -> f16 (50176 x 1536)
  wmma_gemm<0><<<dim3(3 * CCH / 128, MTOK / 128), 128, 0, stream>>>(
      Xw, wqkvT, qkv_b, QKV, nullptr, nullptr, MTOK, 3 * CCH, CCH);

  // windowed multi-head attention -> AttO (f16, windowed token order)
  attention_kernel<<<dim3(NHEADS, BW), 128, 0, stream>>>(QKV, rpbt, AttO);

  // proj + window-reverse + un-shift + residual -> out (f32, original order)
  wmma_gemm<2><<<dim3(CCH / 128, MTOK / 128), 128, 0, stream>>>(
      AttO, wprojT, proj_b, nullptr, out, x, MTOK, CCH, CCH);

  // LN2 -> H2 (f16)
  ln_kernel<false><<<MTOK / 8, 256, 0, stream>>>(out, norm2_s, norm2_b, H2);

  // fc1 + GELU -> Hh (f16, 50176 x 2048)
  wmma_gemm<1><<<dim3(MLPHID / 128, MTOK / 128), 128, 0, stream>>>(
      H2, wfc1T, fc1_b, Hh, nullptr, nullptr, MTOK, MLPHID, CCH);

  // fc2 + residual (in-place on out)
  wmma_gemm<3><<<dim3(CCH / 128, MTOK / 128), 128, 0, stream>>>(
      Hh, wfc2T, fc2_b, nullptr, out, out, MTOK, CCH, MLPHID);
}
